// Scale_Unified_Attention_82454782149066
// MI455X (gfx1250) — matrix operand order repaired and checked by reading, not compiled, not run on hardware
//
#include <hip/hip_runtime.h>
#include <hip/hip_bf16.h>

typedef _Float16 h16;
typedef __attribute__((ext_vector_type(16))) _Float16 v16h;
typedef __attribute__((ext_vector_type(8)))  float    v8f;
typedef __attribute__((ext_vector_type(2)))  float    v2f;
typedef __attribute__((ext_vector_type(4)))  int      i4v;
typedef i4v __attribute__((address_space(1)))* gv4p;  // global int4*
typedef i4v __attribute__((address_space(3)))* lv4p;  // LDS int4*

#ifndef __has_builtin
#define __has_builtin(x) 0
#endif
#if __has_builtin(__builtin_amdgcn_wmma_f32_16x16x4_f32)
#define HAVE_WMMA_F32 1
#else
#define HAVE_WMMA_F32 0
#endif
#if __has_builtin(__builtin_amdgcn_global_load_async_to_lds_b128)
#define HAVE_ASYNC 1
#else
#define HAVE_ASYNC 0
#endif

#define DEV __device__ __forceinline__

DEV float wave_sum32(float v) {
#pragma unroll
  for (int m = 16; m > 0; m >>= 1) v += __shfl_xor(v, m, 32);
  return v;
}

#if HAVE_ASYNC
DEV void async_cp16B(const float* g, float* l) {
  // AS1 address == flat address for global; AS3 address == low 32 bits of
  // flat LDS address.
  __builtin_amdgcn_global_load_async_to_lds_b128(
      (gv4p)(unsigned long long)g,
      (lv4p)(unsigned int)(unsigned long long)l, 0, 0);
}
DEV void wait_async_0() {
#if __has_builtin(__builtin_amdgcn_s_wait_asynccnt)
  __builtin_amdgcn_s_wait_asynccnt(0);
#else
  asm volatile("s_wait_asynccnt 0" ::: "memory");
#endif
}
DEV void wait_async_12() {
#if __has_builtin(__builtin_amdgcn_s_wait_asynccnt)
  __builtin_amdgcn_s_wait_asynccnt(12);
#else
  asm volatile("s_wait_asynccnt 12" ::: "memory");
#endif
}
#endif

// ---------------------------------------------------------------------------
// K1: window_partition + layernorm + sq_lin + qkv  (one wave32 per token)
// writes q/k/v as [win(1024)][head(8)][85][4] fp32 (scale slice at STARTi)
// ---------------------------------------------------------------------------
__global__ __launch_bounds__(128) void k_sq_qkv(
    const float* __restrict__ x, const float* __restrict__ gamma,
    const float* __restrict__ beta, const float* __restrict__ wsq,
    const float* __restrict__ bsq, const float* __restrict__ wqkv,
    const float* __restrict__ bqkv, float* __restrict__ qb,
    float* __restrict__ kb, float* __restrict__ vb, int C, int wsz, int Hdim,
    int SQi, int STARTi) {
  __shared__ float lnb[4][512];
  __shared__ float sqv[4][32];
  int wid = threadIdx.x >> 5, L = threadIdx.x & 31;
  int g = blockIdx.x * 4 + wid;
  int win = g / SQi, tloc = g % SQi;
  int b = win >> 8, p = win & 255;
  int hb = p >> 4, wb = p & 15;
  int y = tloc / wsz, xx = tloc % wsz;
  int row = hb * wsz + y, col = wb * wsz + xx;
  size_t plane = (size_t)Hdim * Hdim;
  const float* xp = x + ((size_t)b * C) * plane + (size_t)row * Hdim + col;
  float s = 0.f;
  for (int c = L; c < C; c += 32) {
    float v = xp[(size_t)c * plane];
    lnb[wid][c] = v;
    s += v;
  }
  float mean = wave_sum32(s) / (float)C;
  float s2 = 0.f;
  for (int c = L; c < C; c += 32) {
    float d = lnb[wid][c] - mean;
    s2 += d * d;
  }
  float rstd = rsqrtf(wave_sum32(s2) / (float)C + 1e-5f);
  for (int c = L; c < C; c += 32)
    lnb[wid][c] = (lnb[wid][c] - mean) * rstd * gamma[c] + beta[c];
  // sq_lin: lane L computes output channel L (of 32)
  float acc = bsq[L];
  for (int c = 0; c < C; ++c) acc += lnb[wid][c] * wsq[c * 32 + L];
  sqv[wid][L] = acc;
  // qkv: lane L computes outputs L, L+32, L+64
  int hh = L >> 2, dd = L & 3;
  size_t dstoff = (((size_t)win * 8 + hh) * 85 + (STARTi + tloc)) * 4 + dd;
#pragma unroll
  for (int part = 0; part < 3; ++part) {
    int o = part * 32 + L;
    float a2 = bqkv[o];
#pragma unroll
    for (int c = 0; c < 32; ++c) a2 += sqv[wid][c] * wqkv[c * 96 + o];
    float* dst = (part == 0) ? qb : (part == 1) ? kb : vb;
    dst[dstoff] = a2;
  }
}

// ---------------------------------------------------------------------------
// K2: per (window, head) attention. S = 0.5*q k^T + bias, softmax, *V
// Uses V_WMMA_F32_16X16X4_F32 (exact fp32) with 6x6 tile grid over 85x85.
// Bias decomposed rel = u(m)+w(n); u/w precomputed in small LDS int tables.
// ---------------------------------------------------------------------------
DEV int off_h85(int q) { return q < 64 ? 0 : q < 80 ? 1 : q < 84 ? 5 : 21; }
DEV int off_w85(int k) { return k < 64 ? 21 : k < 80 ? 5 : k < 84 ? 1 : 0; }

__global__ __launch_bounds__(128) void k_attn(const float* __restrict__ qb,
                                              const float* __restrict__ kb,
                                              const float* __restrict__ vb,
                                              const float* __restrict__ rsb,
                                              float* __restrict__ xa) {
  __shared__ float S[96 * 96];
  __shared__ float vsh[85 * 4];
  __shared__ float bsh[211];
  __shared__ int um[96];
  __shared__ int wn[96];
  int wi = blockIdx.x >> 3, h = blockIdx.x & 7;
  int tid = threadIdx.x, wid = tid >> 5, L = tid & 31;
  const float* qh = qb + (((size_t)wi * 8 + h) * 85) * 4;
  const float* kh = kb + (((size_t)wi * 8 + h) * 85) * 4;
  const float* vh = vb + (((size_t)wi * 8 + h) * 85) * 4;
  for (int i = tid; i < 340; i += 128) vsh[i] = vh[i];
  for (int i = tid; i < 211; i += 128) bsh[i] = rsb[i * 8 + h];
  if (tid < 96) {
    um[tid] = tid + off_h85(tid);
    int w = (84 - tid) + off_w85(tid);
    wn[tid] = w < 0 ? 0 : w;  // negative only for padded n (never read)
  }
  __syncthreads();
#if HAVE_WMMA_F32
  // A layout 16x4 f32: lanes 0-15 rows M=lane K={0,1}; lanes 16-31 K={2,3}
  int k0 = (L >> 4) * 2;
  int lrow = L & 15;
  v2f bf[6];
#pragma unroll
  for (int nt = 0; nt < 6; ++nt) {
    int n = nt * 16 + lrow;
    if (n < 85) {
      bf[nt].x = kh[n * 4 + k0];
      bf[nt].y = kh[n * 4 + k0 + 1];
    } else {
      bf[nt].x = 0.f;
      bf[nt].y = 0.f;
    }
  }
  for (int mt = wid; mt < 6; mt += 4) {  // wave-uniform
    int m0 = mt * 16 + lrow;
    v2f af;
    if (m0 < 85) {
      af.x = qh[m0 * 4 + k0] * 0.5f;  // fold HEAD_DIM^-0.5 = 0.5 into A
      af.y = qh[m0 * 4 + k0 + 1] * 0.5f;
    } else {
      af.x = 0.f;
      af.y = 0.f;
    }
    int mbase = mt * 16 + ((L >> 4) << 3);
#pragma unroll
    for (int nt = 0; nt < 6; ++nt) {
      v8f d = {};
      d = __builtin_amdgcn_wmma_f32_16x16x4_f32(false, af, false, bf[nt],
                                                (short)0, d, false, false);
      int n = nt * 16 + lrow;
      int w = wn[n];  // loop-invariant over r
#pragma unroll
      for (int r = 0; r < 8; ++r) {
        int m = mbase + r;
        int rel = um[m] + w;
        rel = rel > 210 ? 210 : rel;  // only padded (m,n) can exceed
        S[m * 96 + n] = d[r] + bsh[rel];
      }
    }
  }
#else
  for (int idx = tid; idx < 85 * 85; idx += 128) {
    int m = idx / 85, n = idx % 85;
    float a = 0.f;
    for (int k = 0; k < 4; ++k) a += qh[m * 4 + k] * kh[n * 4 + k];
    S[m * 96 + n] = a * 0.5f + bsh[um[m] + wn[n]];
  }
#endif
  __syncthreads();
  if (tid < 85) {
    float mx = -3.0e38f;
    for (int n = 0; n < 85; ++n) mx = fmaxf(mx, S[tid * 96 + n]);
    float se = 0.f;
    for (int n = 0; n < 85; ++n) {
      float e = expf(S[tid * 96 + n] - mx);
      S[tid * 96 + n] = e;
      se += e;
    }
    float inv = 1.0f / se;
    for (int n = 0; n < 85; ++n) S[tid * 96 + n] *= inv;
  }
  __syncthreads();
  float* xo = xa + (((size_t)wi * 8 + h) * 85) * 4;
  for (int idx = tid; idx < 340; idx += 128) {
    int qr = idx >> 2, dd = idx & 3;
    float a = 0.f;
    for (int n = 0; n < 85; ++n) a += S[qr * 96 + n] * vsh[n * 4 + dd];
    xo[idx] = a;
  }
}

// ---------------------------------------------------------------------------
// K3a: scale pooling -> sap -> spq  (one thread per (b, p, head))
// ---------------------------------------------------------------------------
__global__ __launch_bounds__(256) void k_pool(
    const float* __restrict__ xa, const float* __restrict__ sapw,
    const float* __restrict__ sapb, const float* __restrict__ spqw,
    const float* __restrict__ spqb, float* __restrict__ sqh,
    float* __restrict__ skh, float* __restrict__ svb) {
  int id = blockIdx.x * 256 + threadIdx.x;  // 8192
  int b = id >> 11, p = (id >> 3) & 255, h = id & 7;
  int win = b * 256 + p;
  const float* xh = xa + (((size_t)win * 8 + h) * 85) * 4;
  const int starts[5] = {0, 64, 80, 84, 85};
  float m[4][4];
#pragma unroll
  for (int s = 0; s < 4; ++s) {
    float a0 = 0, a1 = 0, a2 = 0, a3 = 0;
    for (int t = starts[s]; t < starts[s + 1]; ++t) {
      a0 += xh[t * 4 + 0];
      a1 += xh[t * 4 + 1];
      a2 += xh[t * 4 + 2];
      a3 += xh[t * 4 + 3];
    }
    float inv = 1.0f / (float)(starts[s + 1] - starts[s]);
    m[s][0] = a0 * inv;
    m[s][1] = a1 * inv;
    m[s][2] = a2 * inv;
    m[s][3] = a3 * inv;
  }
  float r0 = spqb[0], r1 = spqb[1], r2 = spqb[2];
  for (int j = 0; j < 32; ++j) {
#pragma unroll
    for (int d = 0; d < 4; ++d) {
      float pv = sapb[j];
#pragma unroll
      for (int s = 0; s < 4; ++s) pv += m[s][d] * sapw[s * 32 + j];
      int idx = j * 4 + d;
      r0 += pv * spqw[idx * 3 + 0];
      r1 += pv * spqw[idx * 3 + 1];
      r2 += pv * spqw[idx * 3 + 2];
    }
  }
  int o = (b * 8 + h) * 256 + p;
  sqh[o] = r0;
  skh[o] = r1;
  svb[o] = r2;
}

// K3b: depthwise 7x7 LSPE conv on 16x16 grid + residual
__global__ __launch_bounds__(256) void k_conv(const float* __restrict__ svbuf,
                                              const float* __restrict__ lw,
                                              const float* __restrict__ lb,
                                              float* __restrict__ svh) {
  int id = blockIdx.x * 256 + threadIdx.x;
  int b = id >> 11, h = (id >> 8) & 7, y = (id >> 4) & 15, x = id & 15;
  const float* vg = svbuf + (b * 8 + h) * 256;
  float acc = lb[h];
  for (int ky = 0; ky < 7; ++ky) {
    int yy = y + ky - 3;
    if ((unsigned)yy >= 16u) continue;
    for (int kx = 0; kx < 7; ++kx) {
      int xx = x + kx - 3;
      if ((unsigned)xx >= 16u) continue;
      acc += vg[yy * 16 + xx] * lw[h * 49 + ky * 7 + kx];
    }
  }
  svh[id] = vg[y * 16 + x] + acc;
}

// K3c: scalar attention gate: softmax_q(sq[p]*sk[q]) . svh
__global__ __launch_bounds__(256) void k_gate(const float* __restrict__ sqh,
                                              const float* __restrict__ skh,
                                              const float* __restrict__ svh,
                                              float* __restrict__ gate) {
  int id = blockIdx.x * 256 + threadIdx.x;
  int bh = id >> 8, p = id & 255;
  float s = sqh[bh * 256 + p];
  const float* kk = skh + bh * 256;
  const float* vv = svh + bh * 256;
  float mx = -3e38f;
  for (int q2 = 0; q2 < 256; ++q2) mx = fmaxf(mx, s * kk[q2]);
  float se = 0.f, sv = 0.f;
  for (int q2 = 0; q2 < 256; ++q2) {
    float e = expf(s * kk[q2] - mx);
    se += e;
    sv += e * vv[q2];
  }
  gate[id] = sv / se;
}

// ---------------------------------------------------------------------------
// K4: gate-scale + window_reverse + rev LayerNorm (one wave per spatial token)
// ---------------------------------------------------------------------------
__global__ __launch_bounds__(128) void k_gatenorm(
    const float* __restrict__ xa, const float* __restrict__ gate,
    const float* __restrict__ gamma, const float* __restrict__ beta,
    float* __restrict__ Ab, int Hdim, int wsz, int STARTi) {
  int wid = threadIdx.x >> 5, L = threadIdx.x & 31;
  int g = blockIdx.x * 4 + wid;
  int hw = Hdim * Hdim;
  int b = g / hw, rem = g % hw;
  int y = rem / Hdim, x = rem % Hdim;
  int hb = y / wsz, wb = x / wsz;
  int win = b * 256 + hb * 16 + wb;
  int tloc = (y % wsz) * wsz + (x % wsz);
  int h = L >> 2, d = L & 3;
  float val = xa[(((size_t)win * 8 + h) * 85 + STARTi + tloc) * 4 + d] *
              gate[(b * 8 + h) * 256 + hb * 16 + wb];
  float mean = wave_sum32(val) * (1.0f / 32.0f);
  float dv = val - mean;
  float var = wave_sum32(dv * dv) * (1.0f / 32.0f);
  float o = dv * rsqrtf(var + 1e-5f) * gamma[L] + beta[L];
  Ab[(size_t)g * 32 + L] = o;
}

// ---------------------------------------------------------------------------
// WMMA f16 GEMM: out[M,N] = epilogue( A[M,K] @ W[K,N] + bias )
// block = 128 threads (4 waves), tile 128x64, k-step 32, f32 accumulate.
// Each wave owns 32 rows: 2 A-frags x 4 B-subtiles = 8 WMMAs per k-step.
// Async global->LDS copies double-buffered: next k-tile streams in (ASYNCcnt)
// while the current one feeds the WMMAs; s_wait_asynccnt 12 separates stages
// (async loads complete in order per wave).
// ---------------------------------------------------------------------------
__global__ __launch_bounds__(128) void k_gemm(
    const float* __restrict__ A, const float* __restrict__ W,
    const float* __restrict__ bias, const float* __restrict__ res,
    float* __restrict__ out, int M, int N, int K, int act, int mode, int hw) {
  __shared__ float Asf[2][128 * 32];  // [buf][row][k]
  __shared__ float Bsf[2][32 * 64];   // [buf][k][n]
  int tid = threadIdx.x;
  int wid = tid >> 5, L = tid & 31;
  int bm = blockIdx.x, bn = blockIdx.y;
  v8f acc[8] = {v8f{}, v8f{}, v8f{}, v8f{}, v8f{}, v8f{}, v8f{}, v8f{}};

  auto compute = [&](const float* As, const float* Bs) {
    int kbase = (L >> 4) * 16;
    v16h bfr[4];
#pragma unroll
    for (int nt = 0; nt < 4; ++nt) {
      int n = (nt << 4) + (L & 15);
#pragma unroll
      for (int e = 0; e < 16; ++e) bfr[nt][e] = (h16)Bs[(kbase + e) * 64 + n];
    }
#pragma unroll
    for (int mi = 0; mi < 2; ++mi) {
      // A frag: lanes 0-15 row=m, K 0..15; lanes 16-31 K 16..31
      v16h af;
      int row = (wid << 5) + (mi << 4) + (L & 15);
      int base = (L >> 4) * 16;
      const float* p0 = As + row * 32;
#pragma unroll
      for (int e = 0; e < 8; ++e) af[e] = (h16)p0[base + e];
#pragma unroll
      for (int e = 0; e < 8; ++e) af[8 + e] = (h16)p0[base + 8 + e];
#pragma unroll
      for (int nt = 0; nt < 4; ++nt) {
        acc[mi * 4 + nt] = __builtin_amdgcn_wmma_f32_16x16x32_f16(
            false, af, false, bfr[nt], (short)0, acc[mi * 4 + nt], false,
            false);
      }
    }
  };

#if HAVE_ASYNC
  auto issue = [&](int buf, int kt) {
    // A tile: 128x32 f32 = 1024 16B-chunks; 8 per thread
#pragma unroll
    for (int j = 0; j < 8; ++j) {
      int ci = tid + 128 * j;
      int row = ci >> 3, off = (ci & 7) * 4;
      async_cp16B(A + (size_t)(bm * 128 + row) * K + kt + off,
                  Asf[buf] + row * 32 + off);
    }
    // W tile: 32x64 f32 = 512 16B-chunks; 4 per thread
#pragma unroll
    for (int j = 0; j < 4; ++j) {
      int ci = tid + 128 * j;
      int row = ci >> 4, off = (ci & 15) * 4;
      async_cp16B(W + (size_t)(kt + row) * N + bn * 64 + off,
                  Bsf[buf] + row * 64 + off);
    }
  };
  issue(0, 0);
  int cur = 0;
  for (int kt = 0; kt < K; kt += 32) {
    if (kt + 32 < K) {
      issue(1 - cur, kt + 32);  // prefetch next tile into spare buffer
      wait_async_12();          // current tile's 12 copies are done
    } else {
      wait_async_0();
    }
    __syncthreads();
    compute(Asf[cur], Bsf[cur]);
    __syncthreads();  // all reads done before spare buffer refill next iter
    cur ^= 1;
  }
#else
  for (int kt = 0; kt < K; kt += 32) {
    {
      int row = tid;
      const float* src = A + (size_t)(bm * 128 + row) * K + kt;
#pragma unroll
      for (int e = 0; e < 32; ++e) Asf[0][row * 32 + e] = src[e];
    }
    {
      int nn = (tid & 3) * 16;
      int kk2 = tid >> 2;  // 0..31
      const float* src = W + (size_t)(kt + kk2) * N + bn * 64 + nn;
#pragma unroll
      for (int e = 0; e < 16; ++e) Bsf[0][kk2 * 64 + nn + e] = src[e];
    }
    __syncthreads();
    compute(Asf[0], Bsf[0]);
    __syncthreads();
  }
#endif
#pragma unroll
  for (int mi = 0; mi < 2; ++mi) {
#pragma unroll
    for (int nt = 0; nt < 4; ++nt) {
#pragma unroll
      for (int r = 0; r < 8; ++r) {
        int m = bm * 128 + (wid << 5) + (mi << 4) + r + ((L >> 4) << 3);
        int n = bn * 64 + (nt << 4) + (L & 15);
        float v = acc[mi * 4 + nt][r] + bias[n];
        if (act == 1) v = 0.5f * v * (1.0f + erff(v * 0.70710678118654752f));
        if (res) v += res[(size_t)m * N + n];
        if (mode == 0) {
          out[(size_t)m * N + n] = v;
        } else {  // NHWC token-major -> NCHW
          int bI = m / hw;
          int r2 = m % hw;
          out[((size_t)(bI * N + n)) * hw + r2] = v;
        }
      }
    }
  }
}

// ---------------------------------------------------------------------------
// host launcher
// ---------------------------------------------------------------------------
extern "C" void kernel_launch(void* const* d_in, const int* in_sizes, int n_in,
                              void* d_out, int out_size, void* d_ws,
                              size_t ws_size, hipStream_t stream) {
  (void)in_sizes;
  (void)n_in;
  (void)out_size;
  (void)ws_size;
#define IN(i) ((const float*)d_in[(i)])
  const float* x[4] = {IN(0), IN(1), IN(2), IN(3)};
  const float *sq_g[4], *sq_b[4], *sq_w[4], *sq_lb[4], *qkv_w[4], *qkv_b[4];
  const float *rev_g[4], *rev_b[4], *rev_w[4], *rev_lb[4];
  const float *mw1[4], *mb1[4], *mw2[4], *mb2[4];
  for (int i = 0; i < 4; ++i) {
    sq_g[i] = IN(4 + i);
    sq_b[i] = IN(8 + i);
    sq_w[i] = IN(12 + i);
    sq_lb[i] = IN(16 + i);
    qkv_w[i] = IN(20 + i);
    qkv_b[i] = IN(24 + i);
    rev_g[i] = IN(35 + i);
    rev_b[i] = IN(39 + i);
    rev_w[i] = IN(43 + i);
    rev_lb[i] = IN(47 + i);
    mw1[i] = IN(51 + i);
    mb1[i] = IN(55 + i);
    mw2[i] = IN(59 + i);
    mb2[i] = IN(63 + i);
  }
  const float* rsb = IN(28);
  const float* sapw = IN(29);
  const float* sapb = IN(30);
  const float* spqw = IN(31);
  const float* spqb = IN(32);
  const float* lspw = IN(33);
  const float* lspb = IN(34);

  const int DIMS_[4] = {64, 128, 256, 512};
  const int SIZE_[4] = {8, 4, 2, 1};
  const int SQ_[4] = {64, 16, 4, 1};
  const int STARTS_[4] = {0, 64, 80, 84};
  const int Hs[4] = {128, 64, 32, 16};
  const size_t outoff[4] = {0, 4194304, 6291456, 7340032};

  float* ws = (float*)d_ws;
  const size_t QSZ = (size_t)1024 * 8 * 85 * 4;  // 2,785,280
  float* qb = ws;
  float* kb = qb + QSZ;
  float* vb = kb + QSZ;
  float* xa = vb + QSZ;
  float* sqh = xa + QSZ;
  float* skh = sqh + 8192;
  float* svb = skh + 8192;
  float* svh = svb + 8192;
  float* gate = svh + 8192;
  float* Ab = gate + 8192;              // max 65536*32
  float* Pb = Ab + (size_t)65536 * 32;  // max 65536*64
  float* Hb = Pb + (size_t)65536 * 64;  // max 65536*256

  // Stage 1: per-scale LN + sq_lin + qkv into concat q/k/v
  for (int i = 0; i < 4; ++i) {
    int tokens = 1024 * SQ_[i];
    k_sq_qkv<<<tokens / 4, 128, 0, stream>>>(
        x[i], sq_g[i], sq_b[i], sq_w[i], sq_lb[i], qkv_w[i], qkv_b[i], qb, kb,
        vb, DIMS_[i], SIZE_[i], Hs[i], SQ_[i], STARTS_[i]);
  }
  // Stage 2: attention per (window, head)
  k_attn<<<1024 * 8, 128, 0, stream>>>(qb, kb, vb, rsb, xa);
  // Stage 3: scale gating path
  k_pool<<<32, 256, 0, stream>>>(xa, sapw, sapb, spqw, spqb, sqh, skh, svb);
  k_conv<<<32, 256, 0, stream>>>(svb, lspw, lspb, svh);
  k_gate<<<32, 256, 0, stream>>>(sqh, skh, svh, gate);
  // Stage 4: per-scale gate + reverse + LN + proj + MLP (WMMA GEMMs)
  for (int i = 0; i < 4; ++i) {
    int Do = DIMS_[i];
    int tokens = 1024 * SQ_[i];
    int hw = Hs[i] * Hs[i];
    k_gatenorm<<<tokens / 4, 128, 0, stream>>>(xa, gate, rev_g[i], rev_b[i],
                                               Ab, Hs[i], SIZE_[i], STARTS_[i]);
    {  // proj = LN @ rev_lin_w + b   [tokens x 32] @ [32 x Do]
      dim3 grid(tokens / 128, Do / 64);
      k_gemm<<<grid, 128, 0, stream>>>(Ab, rev_w[i], rev_lb[i], nullptr, Pb,
                                       tokens, Do, 32, 0, 0, 0);
    }
    {  // h = gelu(proj @ w1 + b1)    [tokens x Do] @ [Do x 4Do]
      dim3 grid(tokens / 128, (4 * Do) / 64);
      k_gemm<<<grid, 128, 0, stream>>>(Pb, mw1[i], mb1[i], nullptr, Hb, tokens,
                                       4 * Do, Do, 1, 0, 0);
    }
    {  // out = proj + h @ w2 + b2, transposed to NCHW
      dim3 grid(tokens / 128, Do / 64);
      k_gemm<<<grid, 128, 0, stream>>>(Hb, mw2[i], mb2[i], Pb,
                                       (float*)d_out + outoff[i], tokens, Do,
                                       4 * Do, 0, 1, hw);
    }
  }
#undef IN
}
